// TemporalAttention_37099927502934
// MI455X (gfx1250) — matrix operand order repaired and checked by reading, not compiled, not run on hardware
//
#include <hip/hip_runtime.h>

// ---------------------------------------------------------------------------
// Fused dual window cross-attention for MI455X (gfx1250, wave32, WMMA).
// All matmuls: v_wmma_f32_16x16x32_bf16 (f32 accumulate).
// All WMMA operands are stored pre-swizzled in fragment order so each lane
// loads its 16 bf16 elements as one contiguous 32B chunk (2 x b128).
// x1/x2 tiles are staged into LDS by the Tensor Data Mover (6-arg builtin).
// Workspace: ~1.08 GiB.
// ---------------------------------------------------------------------------

#define N_TOK 49
#define NPAD  64
#define CDIM  256
#define NH    8
#define HD    32
#define NB    4096
#define MTOT  (NB * N_TOK)                 // 200704
#define SCALE 0.17677669529663689f         // 32^-0.5

typedef __bf16 bf16_t;
typedef bf16_t v16bf __attribute__((ext_vector_type(16)));
typedef float  v8f   __attribute__((ext_vector_type(8)));
typedef unsigned int u32x4 __attribute__((ext_vector_type(4)));
typedef int i32x8 __attribute__((ext_vector_type(8)));
typedef int i32x4 __attribute__((ext_vector_type(4)));

#if defined(__has_builtin)
#if __has_builtin(__builtin_amdgcn_tensor_load_to_lds) && __has_builtin(__builtin_amdgcn_s_wait_tensorcnt)
#define HAVE_TDM 1
#endif
#endif

static __device__ __forceinline__ unsigned short f2us(float f) {
  unsigned u = __builtin_bit_cast(unsigned, f);
  unsigned r = (u + 0x7FFFu + ((u >> 16) & 1u)) >> 16;   // RNE to bf16
  return (unsigned short)r;
}

// Fragment index maps (contiguous k order k = 16*half + elem, same for A and B).
// A 16x32 bf16, element (m,k):  lane = 16*(k>>4) + m     ; elem = k&15
// B 32x16 bf16, element (k,n):  lane = 16*(k>>4) + n     ; elem = k&15
static __device__ __forceinline__ int a_lane(int m, int k) { return (((k >> 4) & 1) << 4) + (m & 15); }
static __device__ __forceinline__ int a_elem(int k)        { return k & 15; }
static __device__ __forceinline__ int b_lane(int k, int n) { return (((k >> 4) & 1) << 4) + (n & 15); }
static __device__ __forceinline__ int b_elem(int k)        { return k & 15; }

static __device__ __forceinline__ v8f wmma_bf16(v16bf a, v16bf b, v8f c) {
  return __builtin_amdgcn_wmma_f32_16x16x32_bf16(false, a, false, b, (short)0, c,
                                                 false, false);
}
static __device__ __forceinline__ v16bf frag_ld(const unsigned short* base, int frag, int lane) {
  return *(const v16bf*)(base + ((size_t)frag * 32 + lane) * 16);
}

// ---------------------------------------------------------------------------
// Kernel 0a: padded relative-position bias  biasPad[h][row 64][col 64]
// ---------------------------------------------------------------------------
__global__ void k_bias(const int* __restrict__ rel, const float* __restrict__ rpb,
                       float* __restrict__ biasPad) {
  int idx = blockIdx.x * blockDim.x + threadIdx.x;
  if (idx >= NH * NPAD * NPAD) return;
  int h = idx >> 12, rem = idx & 4095, r = rem >> 6, c = rem & 63;
  float v;
  if (c >= N_TOK)       v = -1.0e30f;
  else if (r >= N_TOK)  v = 0.0f;
  else                  v = rpb[rel[r * N_TOK + c] * NH + h];
  biasPad[idx] = v;
}

// ---------------------------------------------------------------------------
// Kernel 0b: prepack one 256x256 f32 weight slab into bf16 B-fragment order.
// B element (k,n) = W[n*256+k]; frag = (n>>4)*8 + (k>>5); 512 elems per frag.
// ---------------------------------------------------------------------------
__global__ void k_packW(const float* __restrict__ W, unsigned short* __restrict__ dst) {
  int t = blockIdx.x * blockDim.x + threadIdx.x;   // 65536 per slab
  int n = t >> 8, k = t & 255;
  int kl = k & 31;
  size_t idx = ((size_t)((n >> 4) * 8 + (k >> 5)) * 32 + b_lane(kl, n)) * 16 + b_elem(kl);
  dst[idx] = f2us(W[t]);
}

// ---------------------------------------------------------------------------
// Kernel 1: per-window QKV projections + cross-head L2-norm epilogue.
// Outputs are written directly in WMMA fragment order:
//   qA  : A-frags (m=token, k=d)          4 frags / (b,h)
//   kB  : B-frags (k=d, n=key token)      4 frags / (b,h)
//   vB  : B-frags (k=key token, n=d)      4 frags / (b,h)
// LDS: sX f32 (2x49x256, TDM dest, phase 0) overlaid with sOut f32 (64x256),
//      then sA bf16 (3 swizzled 64x256 panels). Total 198,656 B.
// ---------------------------------------------------------------------------
__global__ __launch_bounds__(256) void k_qkv(
    const float* __restrict__ x1, const float* __restrict__ x2,
    const unsigned short* __restrict__ wbKV1K, const unsigned short* __restrict__ wbKV1V,
    const unsigned short* __restrict__ wbKV2K, const unsigned short* __restrict__ wbKV2V,
    const unsigned short* __restrict__ wbQ,
    const float* __restrict__ kv1_b, const float* __restrict__ kv2_b,
    const float* __restrict__ q_b,
    unsigned short* __restrict__ qA, float* __restrict__ qraw,
    unsigned short* __restrict__ kB1, unsigned short* __restrict__ vB1,
    unsigned short* __restrict__ kB2, unsigned short* __restrict__ vB2)
{
  extern __shared__ unsigned char smem[];
  float* sOut = (float*)smem;                                // 64KB  (phase B)
  float* sX   = (float*)smem;                                // 98KB  (phase A)
  unsigned short* sA = (unsigned short*)(smem + 100352);     // 3 x 32KB swizzled panels

  const int tid = threadIdx.x;
  const int lane = tid & 31, wv = tid >> 5;
  const int half = lane >> 4, lc = lane & 15;
  const int b = blockIdx.x;

#ifdef HAVE_TDM
  // --- Tensor Data Mover: DMA x1,x2 (49x256 f32 tiles) into LDS -------------
  if (wv == 0) {
    const i32x8 g1 = {0x00020000,              // data_size=4B
                      0x01000000,              // tensor_dim0 = 256 (low16 in [63:48])
                      0x00310000,              // tensor_dim1 = 49  (low16 in [95:80])
                      0x01000000,              // tile_dim0   = 256 ([127:112])
                      49,                      // tile_dim1   = 49  ([143:128])
                      256, 0, 0};              // tensor_dim0_stride = 256
    const i32x4 z4 = {0, 0, 0, 0};
    const i32x8 z8 = {0, 0, 0, 0, 0, 0, 0, 0};
    unsigned ldsBase = (unsigned)(size_t)sX;
#pragma unroll
    for (int s = 0; s < 2; ++s) {
      const float* src = (s == 0) ? x1 : x2;
      unsigned long long ga =
          (unsigned long long)(size_t)(src + (size_t)b * N_TOK * CDIM);
      u32x4 g0 = {1u,                                          // count=1
                  ldsBase + (unsigned)s * (N_TOK * CDIM * 4),  // lds_addr
                  (unsigned)(ga & 0xFFFFFFFFu),
                  (unsigned)((ga >> 32) & 0x01FFFFFFu) | 0x80000000u}; // type=2
      __builtin_amdgcn_tensor_load_to_lds(g0, g1, z4, z4, z8, 0);
    }
    __builtin_amdgcn_s_wait_tensorcnt(0);
  }
  __syncthreads();
#endif

  // Stage bf16 A panels (x1, x2, |x2-x1|) in fragment-swizzled order.
  for (int t = tid; t < NPAD * 128; t += 256) {   // pairs of k
    int m = t >> 7, kp = t & 127, k = kp * 2;
    float a0 = 0.f, a1 = 0.f, c0 = 0.f, c1 = 0.f;
    if (m < N_TOK) {
#ifdef HAVE_TDM
      a0 = sX[m * CDIM + k];     a1 = sX[m * CDIM + k + 1];
      c0 = sX[(N_TOK + m) * CDIM + k]; c1 = sX[(N_TOK + m) * CDIM + k + 1];
#else
      size_t base = ((size_t)b * N_TOK + m) * CDIM + k;
      a0 = x1[base]; a1 = x1[base + 1];
      c0 = x2[base]; c1 = x2[base + 1];
#endif
    }
    int kl = k & 31;
    // even k: elem and elem+1 are adjacent -> store as one b32
    int du = (((m >> 4) * 8 + (k >> 5)) * 32 + a_lane(m, kl)) * 8 + (a_elem(kl) >> 1);
    unsigned pa = (unsigned)f2us(a0) | ((unsigned)f2us(a1) << 16);
    unsigned pc = (unsigned)f2us(c0) | ((unsigned)f2us(c1) << 16);
    unsigned pd = (unsigned)f2us(fabsf(c0 - a0)) | ((unsigned)f2us(fabsf(c1 - a1)) << 16);
    ((unsigned*)sA)[0 * 8192 + du] = pa;
    ((unsigned*)sA)[1 * 8192 + du] = pc;
    ((unsigned*)sA)[2 * 8192 + du] = pd;
  }
  __syncthreads();

  const size_t bhStride = 2048;   // elems per (b,h) in packed q/k/v tensors
  const v8f vz = {0.f, 0.f, 0.f, 0.f, 0.f, 0.f, 0.f, 0.f};

  // chunks: 0=k1 1=v1 2=k2 3=v2 4=q
  for (int chunk = 0; chunk < 5; ++chunk) {
    const unsigned short* Asrc =
        sA + ((chunk < 2) ? 0 : (chunk < 4) ? 16384 : 2 * 16384);
    const unsigned short* Wb;
    const float* biasp;
    switch (chunk) {
      case 0: Wb = wbKV1K; biasp = kv1_b;        break;
      case 1: Wb = wbKV1V; biasp = kv1_b + 256;  break;
      case 2: Wb = wbKV2K; biasp = kv2_b;        break;
      case 3: Wb = wbKV2V; biasp = kv2_b + 256;  break;
      default: Wb = wbQ;   biasp = q_b;          break;
    }

    v8f acc[8];
#pragma unroll
    for (int i = 0; i < 8; ++i) acc[i] = vz;

    for (int kk = 0; kk < 8; ++kk) {           // K = 256 in steps of 32
      v16bf aF[4];
#pragma unroll
      for (int mt = 0; mt < 4; ++mt)
        aF[mt] = frag_ld(Asrc, mt * 8 + kk, lane);
#pragma unroll
      for (int j = 0; j < 2; ++j) {            // wave owns cols [wv*32, wv*32+32)
        v16bf bF = frag_ld(Wb, (wv * 2 + j) * 8 + kk, lane);
#pragma unroll
        for (int mt = 0; mt < 4; ++mt)
          acc[mt * 2 + j] = wmma_bf16(aF[mt], bF, acc[mt * 2 + j]);
      }
    }
    // C + bias -> LDS f32 panel (row-major; epilogue needs cross-head access)
#pragma unroll
    for (int mt = 0; mt < 4; ++mt)
#pragma unroll
      for (int j = 0; j < 2; ++j) {
        int n = wv * 32 + j * 16 + lc;
        float bv = biasp[n];
#pragma unroll
        for (int r = 0; r < 8; ++r) {
          int m = mt * 16 + r + 8 * half;
          sOut[m * CDIM + n] = acc[mt * 2 + j][r] + bv;
        }
      }
    __syncthreads();

    if (chunk == 0 || chunk == 2) {            // k: cross-head L2 norm -> B-frags
      unsigned short* dst = (chunk == 0) ? kB1 : kB2;
      for (int t = tid; t < NPAD * HD; t += 256) {
        int tok = t >> 5, d = t & 31;
        float ss = 0.f;
#pragma unroll
        for (int h = 0; h < NH; ++h) {
          float v = sOut[tok * CDIM + h * HD + d];
          ss += v * v;
        }
        float inv = 1.0f / fmaxf(sqrtf(ss), 1e-12f);
        int fi = (tok >> 4);                       // n-tile
        size_t o = (size_t)(fi * 32 + b_lane(d, tok)) * 16 + b_elem(d);
#pragma unroll
        for (int h = 0; h < NH; ++h)
          dst[(size_t)(b * NH + h) * bhStride + o] =
              f2us(sOut[tok * CDIM + h * HD + d] * inv);
      }
    } else if (chunk == 1 || chunk == 3) {     // v: straight bf16 -> B-frags
      unsigned short* dst = (chunk == 1) ? vB1 : vB2;
      for (int t = tid; t < NPAD * CDIM; t += 256) {
        int tok = t >> 8, cch = t & 255;
        int h = cch >> 5, d = cch & 31;
        int tl = tok & 31;
        int fi = (tok >> 5) * 2 + (d >> 4);        // kh2*2 + dt
        size_t o = (size_t)(fi * 32 + b_lane(tl, d)) * 16 + b_elem(tl);
        dst[(size_t)(b * NH + h) * bhStride + o] = f2us(sOut[t]);
      }
    } else {                                   // q: raw f32 + normalized A-frags
      for (int t = tid; t < NPAD * HD; t += 256) {
        int tok = t >> 5, d = t & 31;
        float ss = 0.f;
#pragma unroll
        for (int h = 0; h < NH; ++h) {
          float v = sOut[tok * CDIM + h * HD + d];
          ss += v * v;
        }
        float inv = 1.0f / fmaxf(sqrtf(ss), 1e-12f);
        size_t o = (size_t)((tok >> 4) * 32 + a_lane(tok, d)) * 16 + a_elem(d);
#pragma unroll
        for (int h = 0; h < NH; ++h) {
          float v = sOut[tok * CDIM + h * HD + d];
          size_t bh = (size_t)(b * NH + h);
          qraw[bh * bhStride + tok * HD + d] = v;   // unscaled
          qA[bh * bhStride + o] = f2us(v * inv);    // L2 norm is scale-invariant
        }
      }
    }
    __syncthreads();
  }
}

// ---------------------------------------------------------------------------
// Kernel 2: attention. One workgroup per window, one wave per head.
// Branch 0 -> k2/v2 (o1, sub q*SCALE); branch 1 -> k1/v1 (o2, sub q*SCALE^2).
// LDS: 128KB staged bias + 64KB fragment-packed softmax strips.
// ---------------------------------------------------------------------------
__global__ __launch_bounds__(256) void k_attn(
    const unsigned short* __restrict__ qA, const float* __restrict__ qraw,
    const unsigned short* __restrict__ kB1, const unsigned short* __restrict__ vB1,
    const unsigned short* __restrict__ kB2, const unsigned short* __restrict__ vB2,
    const float* __restrict__ biasPad,
    unsigned short* __restrict__ o1pre, unsigned short* __restrict__ o2pre)
{
  extern __shared__ unsigned char smem[];
  float* sBias = (float*)smem;                                   // 8 * 4096 f32
  unsigned short* sP = (unsigned short*)(smem + NH * 4096 * 4);  // 8 * 4096 bf16

  const int tid = threadIdx.x, lane = tid & 31, wv = tid >> 5;
  const int half = lane >> 4, lc = lane & 15;
  const int b = blockIdx.x, h = wv;

  float* biasW = sBias + wv * 4096;
  for (int i = lane; i < 4096; i += 32) biasW[i] = biasPad[h * 4096 + i];
  unsigned short* P = sP + wv * 4096;          // 8 A-frags (4 mt x 2 khalf)

  const size_t headBase = (size_t)(b * NH + h) * 2048;
  const float* qrh = qraw + headBase;

  v16bf aQ[4];
#pragma unroll
  for (int mt = 0; mt < 4; ++mt) aQ[mt] = frag_ld(qA + headBase, mt, lane);
  __syncthreads();

  const v8f vz = {0.f, 0.f, 0.f, 0.f, 0.f, 0.f, 0.f, 0.f};

  for (int br = 0; br < 2; ++br) {
    const unsigned short* kh = (br == 0 ? kB2 : kB1) + headBase;
    const unsigned short* vh = (br == 0 ? vB2 : vB1) + headBase;
    unsigned short* op = (br == 0) ? o1pre : o2pre;
    const float subScale = (br == 0) ? SCALE : (SCALE * SCALE);

    // S = qn @ kn^T : 64x64 (16 tiles), K=32 single WMMA
    v8f S[16];
#pragma unroll
    for (int i = 0; i < 16; ++i) S[i] = vz;
#pragma unroll
    for (int nt = 0; nt < 4; ++nt) {
      v16bf bK = frag_ld(kh, nt, lane);
#pragma unroll
      for (int mt = 0; mt < 4; ++mt)
        S[mt * 4 + nt] = wmma_bf16(aQ[mt], bK, S[mt * 4 + nt]);
    }

    // bias + row softmax; write probs straight into A-fragment order
#pragma unroll
    for (int mt = 0; mt < 4; ++mt)
#pragma unroll
      for (int r = 0; r < 8; ++r) {
        int m = mt * 16 + r + 8 * half;
        float sv[4];
        float mx = -3.0e38f;
#pragma unroll
        for (int nt = 0; nt < 4; ++nt) {
          float x = S[mt * 4 + nt][r] + biasW[m * 64 + nt * 16 + lc];
          sv[nt] = x;
          mx = fmaxf(mx, x);
        }
#pragma unroll
        for (int d = 1; d < 16; d <<= 1) mx = fmaxf(mx, __shfl_xor(mx, d, 16));
        float sum = 0.f;
#pragma unroll
        for (int nt = 0; nt < 4; ++nt) { sv[nt] = __expf(sv[nt] - mx); sum += sv[nt]; }
#pragma unroll
        for (int d = 1; d < 16; d <<= 1) sum += __shfl_xor(sum, d, 16);
        float inv = 1.0f / sum;
#pragma unroll
        for (int nt = 0; nt < 4; ++nt) {
          int col = nt * 16 + lc, cl = col & 31;
          int fi = mt * 2 + (nt >> 1);
          P[(fi * 32 + a_lane(m, cl)) * 16 + a_elem(cl)] = f2us(sv[nt] * inv);
        }
      }
    __syncthreads();

    // O = P @ v : 64x32 (8 tiles), K=64 in two WMMA steps
    v8f O[8];
#pragma unroll
    for (int i = 0; i < 8; ++i) O[i] = vz;
#pragma unroll
    for (int kh2 = 0; kh2 < 2; ++kh2) {
      v16bf aP[4];
#pragma unroll
      for (int mt = 0; mt < 4; ++mt) aP[mt] = frag_ld(P, mt * 2 + kh2, lane);
#pragma unroll
      for (int dt = 0; dt < 2; ++dt) {
        v16bf bV = frag_ld(vh, kh2 * 2 + dt, lane);
#pragma unroll
        for (int mt = 0; mt < 4; ++mt)
          O[mt * 2 + dt] = wmma_bf16(aP[mt], bV, O[mt * 2 + dt]);
      }
    }

    // epilogue: subtract scaled q, write bf16 [token, channel]
#pragma unroll
    for (int mt = 0; mt < 4; ++mt)
#pragma unroll
      for (int dt = 0; dt < 2; ++dt)
#pragma unroll
        for (int r = 0; r < 8; ++r) {
          int m = mt * 16 + r + 8 * half;
          if (m < N_TOK) {
            int d = dt * 16 + lc;
            float val = O[mt * 2 + dt][r] - qrh[m * HD + d] * subScale;
            op[(size_t)(b * N_TOK + m) * CDIM + h * HD + d] = f2us(val);
          }
        }
    __syncthreads();
  }
}

// ---------------------------------------------------------------------------
// Kernel 3: output projections (two branches via grid.y), fp32 result.
// 64 rows x 256 cols per workgroup; A staged swizzled in 32KB LDS.
// ---------------------------------------------------------------------------
__global__ __launch_bounds__(256) void k_proj(
    const unsigned short* __restrict__ o1pre, const unsigned short* __restrict__ o2pre,
    const unsigned short* __restrict__ wbP1, const unsigned short* __restrict__ wbP2,
    const float* __restrict__ b1, const float* __restrict__ b2,
    float* __restrict__ out)
{
  extern __shared__ unsigned char smem[];
  unsigned short* sA = (unsigned short*)smem;   // 64x256 bf16, fragment order

  const int tid = threadIdx.x, lane = tid & 31, wv = tid >> 5;
  const int half = lane >> 4, lc = lane & 15;
  const int mb = blockIdx.x, br = blockIdx.y;

  const unsigned short* src = br ? o2pre : o1pre;
  const unsigned short* Wb = br ? wbP2 : wbP1;
  const float* bias = br ? b2 : b1;
  float* outp = out + (size_t)br * MTOT * CDIM;

  // swizzle-stage A: read row-major pairs, write fragment-order b32
  const unsigned int* s32 = (const unsigned int*)(src + (size_t)mb * 64 * CDIM);
  unsigned int* d32 = (unsigned int*)sA;
  for (int t = tid; t < 64 * 128; t += 256) {
    int m = t >> 7, k = (t & 127) * 2;
    int kl = k & 31;
    int du = (((m >> 4) * 8 + (k >> 5)) * 32 + a_lane(m, kl)) * 8 + (a_elem(kl) >> 1);
    d32[du] = s32[t];
  }
  __syncthreads();

  const v8f vz = {0.f, 0.f, 0.f, 0.f, 0.f, 0.f, 0.f, 0.f};
  v8f acc[8];
#pragma unroll
  for (int i = 0; i < 8; ++i) acc[i] = vz;

  for (int kk = 0; kk < 8; ++kk) {
    v16bf aF[4];
#pragma unroll
    for (int mt = 0; mt < 4; ++mt)
      aF[mt] = frag_ld(sA, mt * 8 + kk, lane);
#pragma unroll
    for (int j = 0; j < 2; ++j) {
      v16bf bF = frag_ld(Wb, (wv * 2 + j) * 8 + kk, lane);
#pragma unroll
      for (int mt = 0; mt < 4; ++mt)
        acc[mt * 2 + j] = wmma_bf16(aF[mt], bF, acc[mt * 2 + j]);
    }
  }
#pragma unroll
  for (int mt = 0; mt < 4; ++mt)
#pragma unroll
    for (int j = 0; j < 2; ++j) {
      int n = wv * 32 + j * 16 + lc;
      float bv = bias[n];
#pragma unroll
      for (int r = 0; r < 8; ++r) {
        int m = mt * 16 + r + 8 * half;
        outp[(size_t)(mb * 64 + m) * CDIM + n] = acc[mt * 2 + j][r] + bv;
      }
    }
}

// ---------------------------------------------------------------------------
extern "C" void kernel_launch(void* const* d_in, const int* in_sizes, int n_in,
                              void* d_out, int out_size, void* d_ws, size_t ws_size,
                              hipStream_t stream) {
  (void)in_sizes; (void)n_in; (void)out_size; (void)ws_size;

  const float* x1    = (const float*)d_in[0];
  const float* x2    = (const float*)d_in[1];
  const int*   rel   = (const int*)d_in[2];
  const float* kv1_w = (const float*)d_in[3];
  const float* kv1_b = (const float*)d_in[4];
  const float* kv2_w = (const float*)d_in[5];
  const float* kv2_b = (const float*)d_in[6];
  const float* q_w   = (const float*)d_in[7];
  const float* q_b   = (const float*)d_in[8];
  const float* rpb   = (const float*)d_in[9];
  const float* p1w   = (const float*)d_in[10];
  const float* p1b   = (const float*)d_in[11];
  const float* p2w   = (const float*)d_in[12];
  const float* p2b   = (const float*)d_in[13];

  unsigned char* ws = (unsigned char*)d_ws;
  const size_t EHD = (size_t)NB * NH * 2048;   // 67,108,864 elements
  size_t off = 0;
  unsigned short* qAp = (unsigned short*)(ws + off); off += EHD * 2;
  unsigned short* kB1 = (unsigned short*)(ws + off); off += EHD * 2;
  unsigned short* vB1 = (unsigned short*)(ws + off); off += EHD * 2;
  unsigned short* kB2 = (unsigned short*)(ws + off); off += EHD * 2;
  unsigned short* vB2 = (unsigned short*)(ws + off); off += EHD * 2;
  float*          qrw = (float*)(ws + off);          off += EHD * 4;
  unsigned short* o1p = (unsigned short*)(ws + off); off += (size_t)MTOT * CDIM * 2;
  unsigned short* o2p = (unsigned short*)(ws + off); off += (size_t)MTOT * CDIM * 2;
  float*          bpd = (float*)(ws + off);          off += (size_t)NH * NPAD * NPAD * 4;
  unsigned short* wSlab[7];
  for (int i = 0; i < 7; ++i) { wSlab[i] = (unsigned short*)(ws + off); off += 65536 * 2; }
  // requires ws_size >= off (~1.08 GiB)

  k_bias<<<dim3((NH * NPAD * NPAD + 255) / 256), dim3(256), 0, stream>>>(rel, rpb, bpd);

  const float* slabSrc[7] = {kv1_w, kv1_w + 65536, kv2_w, kv2_w + 65536, q_w, p1w, p2w};
  for (int i = 0; i < 7; ++i)
    k_packW<<<dim3(256), dim3(256), 0, stream>>>(slabSrc[i], wSlab[i]);

  k_qkv<<<dim3(NB), dim3(256), 198656, stream>>>(
      x1, x2, wSlab[0], wSlab[1], wSlab[2], wSlab[3], wSlab[4],
      kv1_b, kv2_b, q_b, qAp, qrw, kB1, vB1, kB2, vB2);

  k_attn<<<dim3(NB), dim3(256), NH * 4096 * 4 + NH * 4096 * 2, stream>>>(
      qAp, qrw, kB1, vB1, kB2, vB2, bpd, o1p, o2p);

  k_proj<<<dim3(MTOT / 64, 2), dim3(256), 64 * CDIM * 2, stream>>>(
      o1p, o2p, wSlab[5], wSlab[6], p1b, p2b, (float*)d_out);
}
